// GPTMultiHeadAttention_78194174591312
// MI455X (gfx1250) — matrix operand order repaired and checked by reading, not compiled, not run on hardware
//
#include <hip/hip_runtime.h>
#include <hip/hip_bf16.h>

// ---------------------------------------------------------------------------
// GPT multi-head attention forward for gfx1250 (MI455X), wave32 + WMMA bf16
// + double-buffered Tensor Data Mover staging of K/V tiles into LDS.
// B=2, T=2048, C=1024, H=16, d=64.
// ---------------------------------------------------------------------------

typedef __attribute__((ext_vector_type(16))) __bf16 v16bf;
typedef __attribute__((ext_vector_type(8)))  float  v8f;
typedef __attribute__((ext_vector_type(4)))  unsigned int v4u;
typedef __attribute__((ext_vector_type(8)))  int v8i;
typedef __attribute__((ext_vector_type(4)))  int v4i;

#define BB 2
#define TT 2048
#define CC 1024
#define HH 16
#define DD 64

#if defined(__gfx1250__) && __has_builtin(__builtin_amdgcn_tensor_load_to_lds)
#define USE_TDM 1
#else
#define USE_TDM 0
#endif

__device__ __forceinline__ v8f wmma_bf16(v16bf a, v16bf b, v8f c) {
  return __builtin_amdgcn_wmma_f32_16x16x32_bf16(
      /*neg_a=*/false, a, /*neg_b=*/false, b,
      /*c_mod=*/(short)0, c, /*reuse_a=*/false, /*reuse_b=*/false);
}

// A fragment (16x32, 16-bit): lane<16 holds M=lane, K = {0..7, 16..23};
// lane>=16 holds M=lane-16, K = {8..15, 24..31}.  src is row-major [M, ld].
__device__ __forceinline__ v16bf load_a_f32(const float* __restrict__ src,
                                            int ld, int lane) {
  const int m  = lane & 15;
  const int kh = (lane >> 4) & 1;
  const float* row = src + m * ld;
  v16bf a;
#pragma unroll
  for (int g = 0; g < 2; ++g) {
    const int kbase = g * 16 + kh * 8;
#pragma unroll
    for (int i = 0; i < 8; ++i) a[g * 8 + i] = (__bf16)row[kbase + i];
  }
  return a;
}

__device__ __forceinline__ v16bf load_a_bf16(const __bf16* src, int ld, int lane) {
  const int m  = lane & 15;
  const int kh = (lane >> 4) & 1;
  const __bf16* row = src + m * ld;
  v16bf a;
#pragma unroll
  for (int g = 0; g < 2; ++g) {
    const int kbase = g * 16 + kh * 8;
#pragma unroll
    for (int i = 0; i < 8; ++i) a[g * 8 + i] = row[kbase + i];
  }
  return a;
}

// B fragment (32x16, 16-bit): lane = column N (lane&15); lanes 0-15 cover
// K = {0..7, 16..23}, lanes 16-31 cover K = {8..15, 24..31} (same as A).
// Source layout: value(k, n) = src[n*ld + k]  (K-contiguous per lane).
__device__ __forceinline__ v16bf load_b_kcontig(const __bf16* src, int ld, int lane) {
  const int n  = lane & 15;
  const int kh = (lane >> 4) & 1;
  const __bf16* col = src + n * ld;
  v16bf b;
#pragma unroll
  for (int i = 0; i < 16; ++i) b[i] = col[(i >> 3) * 16 + kh * 8 + (i & 7)];
  return b;
}

#if USE_TDM
// Issue a TDM 2D tile load (bf16 elements) global -> LDS.
// tile_d0 = contiguous elements per row, tile_d1 = rows,
// stride0 = row stride in elements.  Tensor dims == tile dims (tile in-range).
__device__ __forceinline__ void tdm_load_2d(unsigned lds_off, const void* gptr,
                                            unsigned tile_d0, unsigned tile_d1,
                                            unsigned long long stride0) {
  const unsigned long long ga = (unsigned long long)(uintptr_t)gptr;
  v4u g0;
  g0[0] = 1u;                                       // count=1, user descriptor
  g0[1] = lds_off;                                  // lds_addr [63:32]
  g0[2] = (unsigned)(ga & 0xFFFFFFFFu);             // global_addr lo
  g0[3] = (unsigned)((ga >> 32) & 0x01FFFFFFu) | (2u << 30);  // hi | type=2
  v8i g1;
  g1[0] = (int)(1u << 16);                          // data_size=1 (2B), mask=0
  g1[1] = (int)((tile_d0 & 0xFFFFu) << 16);         // tensor_dim0[15:0]
  g1[2] = (int)((tile_d0 >> 16) | ((tile_d1 & 0xFFFFu) << 16)); // td0 hi|td1 lo
  g1[3] = (int)((tile_d1 >> 16) | ((tile_d0 & 0xFFFFu) << 16)); // td1 hi|tile_dim0
  g1[4] = (int)(tile_d1 & 0xFFFFu);                 // tile_dim1 | tile_dim2=0
  g1[5] = (int)(stride0 & 0xFFFFFFFFull);           // tensor_dim0_stride lo
  g1[6] = (int)((stride0 >> 32) & 0xFFFFull);       // stride hi | stride1=0
  g1[7] = 0;
  v4i gz = {0, 0, 0, 0};
#if __clang_major__ >= 23
  v8i gz8 = {0, 0, 0, 0, 0, 0, 0, 0};
  __builtin_amdgcn_tensor_load_to_lds(g0, g1, gz, gz, gz8, 0);
#else
  __builtin_amdgcn_tensor_load_to_lds(g0, g1, gz, gz, 0);
#endif
}
#endif

// ---------------------------------------------------------------------------
// Kernel 0: transpose + convert a 1024x1024 fp32 weight to bf16 [N, K].
// ---------------------------------------------------------------------------
__global__ void k_transpose_cvt(const float* __restrict__ w,
                                __bf16* __restrict__ wt) {
  __shared__ float tile[32][33];
  const int bx = blockIdx.x * 32;   // K base
  const int by = blockIdx.y * 32;   // N base
  const int tx = threadIdx.x & 31;
  const int ty = threadIdx.x >> 5;  // 0..7
#pragma unroll
  for (int i = ty; i < 32; i += 8)
    tile[i][tx] = w[(bx + i) * CC + by + tx];   // tile[k][n]
  __syncthreads();
#pragma unroll
  for (int i = ty; i < 32; i += 8)
    wt[(by + i) * CC + bx + tx] = (__bf16)tile[tx][i];  // wt[n][k]
}

// ---------------------------------------------------------------------------
// Kernel 1: fused QKV projection.  Each wave computes a 32x64 output strip
// (2 A frags, 4 B frags, 8 WMMAs per k-step -> independent loads for overlap).
//   z=0: qb [B,H,T,d] (scaled by 1/sqrt(d)), z=1: kb [B,H,T,d],
//   z=2: vtb [B,H,d,T] (transposed).
// ---------------------------------------------------------------------------
__global__ void k_qkv(const float* __restrict__ x,
                      const __bf16* __restrict__ wqt,
                      const __bf16* __restrict__ wkt,
                      const __bf16* __restrict__ wvt,
                      const float* __restrict__ bq,
                      const float* __restrict__ bk,
                      const float* __restrict__ bv,
                      __bf16* __restrict__ qb,
                      __bf16* __restrict__ kb,
                      __bf16* __restrict__ vtb) {
  const int lane = threadIdx.x & 31;
  const int wave = threadIdx.x >> 5;
  const int m0 = blockIdx.x * 128 + wave * 32;
  const int n0 = blockIdx.y * 64;
  const int z  = blockIdx.z;
  const __bf16* wt  = (z == 0) ? wqt : (z == 1) ? wkt : wvt;
  const float*  bia = (z == 0) ? bq  : (z == 1) ? bk  : bv;

  const v8f vzero = {0.f, 0.f, 0.f, 0.f, 0.f, 0.f, 0.f, 0.f};
  v8f c[8] = {vzero, vzero, vzero, vzero, vzero, vzero, vzero, vzero};

  for (int k0 = 0; k0 < CC; k0 += 32) {
    v16bf bfr[4];
#pragma unroll
    for (int j = 0; j < 4; ++j)
      bfr[j] = load_b_kcontig(wt + (n0 + 16 * j) * CC + k0, CC, lane);
    v16bf a0 = load_a_f32(x + m0 * CC + k0, CC, lane);
    v16bf a1 = load_a_f32(x + (m0 + 16) * CC + k0, CC, lane);
    if (k0 + 32 < CC) {
      __builtin_prefetch(x + m0 * CC + k0 + 32, 0, 0);
      __builtin_prefetch(wt + n0 * CC + k0 + 32, 0, 0);
    }
#pragma unroll
    for (int j = 0; j < 4; ++j) c[j]     = wmma_bf16(a0, bfr[j], c[j]);
#pragma unroll
    for (int j = 0; j < 4; ++j) c[4 + j] = wmma_bf16(a1, bfr[j], c[4 + j]);
  }

  const int ncol  = lane & 15;
  const int mbase = (lane >> 4) * 8;
#pragma unroll
  for (int j = 0; j < 4; ++j) {
    const int gc = n0 + 16 * j + ncol;   // column in [0, C)
    const float bval = bia[gc];
    const int h  = gc >> 6;              // head
    const int dd = gc & 63;              // dim within head
#pragma unroll
    for (int half = 0; half < 2; ++half) {
#pragma unroll
      for (int r = 0; r < 8; ++r) {
        const int gm = m0 + half * 16 + mbase + r;  // global token index
        const int bi = gm >> 11;                    // batch (T = 2048)
        const int t  = gm & 2047;
        const float val = c[half * 4 + j][r] + bval;
        const int bh = bi * HH + h;
        if (z == 0)
          qb[((bh * TT + t) << 6) + dd] = (__bf16)(val * 0.125f);  // 1/sqrt(64)
        else if (z == 1)
          kb[((bh * TT + t) << 6) + dd] = (__bf16)val;
        else
          vtb[((bh << 6) + dd) * TT + t] = (__bf16)val;
      }
    }
  }
}

// ---------------------------------------------------------------------------
// Kernel 2: causal flash attention.  One wave per (b, h, 16-row Q block).
// K/V tiles are staged to LDS by the Tensor Data Mover with 2-deep double
// buffering: issue step s+1's loads, s_wait_tensorcnt 2 (in-order completion
// guarantees step s's tiles are resident), compute step s.
// ---------------------------------------------------------------------------
__global__ void __launch_bounds__(32)
k_attn(const __bf16* __restrict__ qb,
       const __bf16* __restrict__ kb,
       const __bf16* __restrict__ vtb,
       float* __restrict__ attn_out) {
  __shared__ __bf16 p_lds[16 * 32];
#if USE_TDM
  __shared__ __bf16 k_tile[2][32 * 64];   // [buf][token][dd]
  __shared__ __bf16 v_tile[2][64 * 32];   // [buf][dd][token]
#endif

  const int lane = threadIdx.x;
  const int q0 = blockIdx.x * 16;
  const int bh = blockIdx.y;            // b*H + h
  const int bi = bh >> 4;
  const int h  = bh & 15;

  const __bf16* qbase = qb  + bh * TT * DD;
  const __bf16* kbase = kb  + bh * TT * DD;
  const __bf16* vbase = vtb + bh * DD * TT;

  const v16bf qa0 = load_a_bf16(qbase + q0 * DD,      DD, lane);
  const v16bf qa1 = load_a_bf16(qbase + q0 * DD + 32, DD, lane);

  const v8f vzero = {0.f, 0.f, 0.f, 0.f, 0.f, 0.f, 0.f, 0.f};
  v8f oc[4] = {vzero, vzero, vzero, vzero};
  float run_max[8], run_sum[8];
#pragma unroll
  for (int r = 0; r < 8; ++r) { run_max[r] = -3.0e38f; run_sum[r] = 0.0f; }

  const int ncol  = lane & 15;
  const int mbase = (lane >> 4) * 8;
  const int nsteps = (q0 + 47) >> 5;    // cover columns [0, q0+16) in 32s

#if USE_TDM
  // Prologue: stage step 0 into buffer 0.
  tdm_load_2d((unsigned)(uintptr_t)&k_tile[0][0], kbase,
              /*d0=*/DD, /*d1=*/32, /*stride=*/DD);
  tdm_load_2d((unsigned)(uintptr_t)&v_tile[0][0], vbase,
              /*d0=*/32, /*d1=*/DD, /*stride=*/TT);
#endif

  for (int s = 0; s < nsteps; ++s) {
    const int kb0 = s * 32;
    const int cur = s & 1;
    (void)cur;

    // DS reads from the buffer about to be overwritten (used at step s-1)
    // must be complete; the barrier's fence forces s_wait_dscnt.
    __syncthreads();

    v8f s0 = vzero, s1 = vzero;
#if USE_TDM
    if (s + 1 < nsteps) {
      const int kn = (s + 1) * 32;
      tdm_load_2d((unsigned)(uintptr_t)&k_tile[1 - cur][0], kbase + kn * DD,
                  /*d0=*/DD, /*d1=*/32, /*stride=*/DD);
      tdm_load_2d((unsigned)(uintptr_t)&v_tile[1 - cur][0], vbase + kn,
                  /*d0=*/32, /*d1=*/DD, /*stride=*/TT);
      __builtin_amdgcn_s_wait_tensorcnt(2);   // step s's tiles resident
    } else {
      __builtin_amdgcn_s_wait_tensorcnt(0);
    }
    const __bf16* kt = &k_tile[cur][0];
    const __bf16* vt = &v_tile[cur][0];
    {
      v16bf b;
      b = load_b_kcontig(kt,                DD, lane); s0 = wmma_bf16(qa0, b, s0);
      b = load_b_kcontig(kt + 32,           DD, lane); s0 = wmma_bf16(qa1, b, s0);
      b = load_b_kcontig(kt + 16 * DD,      DD, lane); s1 = wmma_bf16(qa0, b, s1);
      b = load_b_kcontig(kt + 16 * DD + 32, DD, lane); s1 = wmma_bf16(qa1, b, s1);
    }
#else
    {
      v16bf b;
      b = load_b_kcontig(kbase + kb0 * DD,             DD, lane); s0 = wmma_bf16(qa0, b, s0);
      b = load_b_kcontig(kbase + kb0 * DD + 32,        DD, lane); s0 = wmma_bf16(qa1, b, s0);
      b = load_b_kcontig(kbase + (kb0 + 16) * DD,      DD, lane); s1 = wmma_bf16(qa0, b, s1);
      b = load_b_kcontig(kbase + (kb0 + 16) * DD + 32, DD, lane); s1 = wmma_bf16(qa1, b, s1);
    }
#endif

    // ---- causal mask ----
#pragma unroll
    for (int r = 0; r < 8; ++r) {
      const int qrow = q0 + mbase + r;
      const int kc   = kb0 + ncol;
      s0[r] = (kc      > qrow) ? -1.0e30f : s0[r];
      s1[r] = (kc + 16 > qrow) ? -1.0e30f : s1[r];
    }

    // ---- online softmax (row stats across each 16-lane half-wave) ----
#pragma unroll
    for (int r = 0; r < 8; ++r) {
      float rm = fmaxf(s0[r], s1[r]);
#pragma unroll
      for (int off = 1; off < 16; off <<= 1) rm = fmaxf(rm, __shfl_xor(rm, off));
      const float nm    = fmaxf(run_max[r], rm);
      const float alpha = __expf(run_max[r] - nm);
      const float p0 = __expf(s0[r] - nm);
      const float p1 = __expf(s1[r] - nm);
      float ps = p0 + p1;
#pragma unroll
      for (int off = 1; off < 16; off <<= 1) ps += __shfl_xor(ps, off);
      run_sum[r] = run_sum[r] * alpha + ps;
      run_max[r] = nm;
#pragma unroll
      for (int j = 0; j < 4; ++j) oc[j][r] *= alpha;
      p_lds[(mbase + r) * 32 + ncol]      = (__bf16)p0;
      p_lds[(mbase + r) * 32 + ncol + 16] = (__bf16)p1;
    }
    __syncthreads();

    // ---- P @ V ----
    const v16bf pa = load_a_bf16(p_lds, 32, lane);
#pragma unroll
    for (int j = 0; j < 4; ++j) {
#if USE_TDM
      v16bf vb = load_b_kcontig(vt + (16 * j) * 32, 32, lane);
#else
      v16bf vb = load_b_kcontig(vbase + (16 * j) * TT + kb0, TT, lane);
#endif
      oc[j] = wmma_bf16(pa, vb, oc[j]);
    }
  }

  // ---- normalize and write merged-head fp32 output [B, T, C] ----
#pragma unroll
  for (int j = 0; j < 4; ++j) {
#pragma unroll
    for (int r = 0; r < 8; ++r) {
      const int t   = q0 + mbase + r;
      const int col = h * DD + 16 * j + ncol;
      attn_out[(bi * TT + t) * CC + col] = oc[j][r] / run_sum[r];
    }
  }
}

// ---------------------------------------------------------------------------
// Kernel 3: output projection  out = attn @ wo + bo  (fp32 out), 32x64/wave.
// ---------------------------------------------------------------------------
__global__ void k_oproj(const float* __restrict__ attn,
                        const __bf16* __restrict__ wot,
                        const float* __restrict__ bo,
                        float* __restrict__ out) {
  const int lane = threadIdx.x & 31;
  const int wave = threadIdx.x >> 5;
  const int m0 = blockIdx.x * 128 + wave * 32;
  const int n0 = blockIdx.y * 64;

  const v8f vzero = {0.f, 0.f, 0.f, 0.f, 0.f, 0.f, 0.f, 0.f};
  v8f c[8] = {vzero, vzero, vzero, vzero, vzero, vzero, vzero, vzero};

  for (int k0 = 0; k0 < CC; k0 += 32) {
    v16bf bfr[4];
#pragma unroll
    for (int j = 0; j < 4; ++j)
      bfr[j] = load_b_kcontig(wot + (n0 + 16 * j) * CC + k0, CC, lane);
    v16bf a0 = load_a_f32(attn + m0 * CC + k0, CC, lane);
    v16bf a1 = load_a_f32(attn + (m0 + 16) * CC + k0, CC, lane);
    if (k0 + 32 < CC) {
      __builtin_prefetch(attn + m0 * CC + k0 + 32, 0, 0);
      __builtin_prefetch(wot + n0 * CC + k0 + 32, 0, 0);
    }
#pragma unroll
    for (int j = 0; j < 4; ++j) c[j]     = wmma_bf16(a0, bfr[j], c[j]);
#pragma unroll
    for (int j = 0; j < 4; ++j) c[4 + j] = wmma_bf16(a1, bfr[j], c[4 + j]);
  }

  const int ncol  = lane & 15;
  const int mbase = (lane >> 4) * 8;
#pragma unroll
  for (int j = 0; j < 4; ++j) {
    const int gc = n0 + 16 * j + ncol;
    const float bval = bo[gc];
#pragma unroll
    for (int half = 0; half < 2; ++half)
#pragma unroll
      for (int r = 0; r < 8; ++r)
        out[(m0 + half * 16 + mbase + r) * CC + gc] = c[half * 4 + j][r] + bval;
  }
}

// ---------------------------------------------------------------------------
extern "C" void kernel_launch(void* const* d_in, const int* in_sizes, int n_in,
                              void* d_out, int out_size, void* d_ws, size_t ws_size,
                              hipStream_t stream) {
  (void)in_sizes; (void)n_in; (void)out_size; (void)ws_size;

  const float* x  = (const float*)d_in[0];
  const float* wq = (const float*)d_in[1];
  const float* bq = (const float*)d_in[2];
  const float* wk = (const float*)d_in[3];
  const float* bk = (const float*)d_in[4];
  const float* wv = (const float*)d_in[5];
  const float* bv = (const float*)d_in[6];
  const float* wo = (const float*)d_in[7];
  const float* bo = (const float*)d_in[8];
  float* out = (float*)d_out;

  char* p = (char*)d_ws;
  const size_t wbytes = (size_t)CC * CC * sizeof(__bf16);           // 2 MB
  const size_t qbytes = (size_t)BB * HH * TT * DD * sizeof(__bf16); // 8 MB
  __bf16* wqt = (__bf16*)p; p += wbytes;
  __bf16* wkt = (__bf16*)p; p += wbytes;
  __bf16* wvt = (__bf16*)p; p += wbytes;
  __bf16* wot = (__bf16*)p; p += wbytes;
  __bf16* qb  = (__bf16*)p; p += qbytes;
  __bf16* kb  = (__bf16*)p; p += qbytes;
  __bf16* vtb = (__bf16*)p; p += qbytes;
  float* attn = (float*)p;  p += (size_t)BB * TT * CC * sizeof(float); // 16 MB

  dim3 tg(32, 32);
  k_transpose_cvt<<<tg, 256, 0, stream>>>(wq, wqt);
  k_transpose_cvt<<<tg, 256, 0, stream>>>(wk, wkt);
  k_transpose_cvt<<<tg, 256, 0, stream>>>(wv, wvt);
  k_transpose_cvt<<<tg, 256, 0, stream>>>(wo, wot);

  k_qkv<<<dim3(BB * TT / 128, CC / 64, 3), 128, 0, stream>>>(
      x, wqt, wkt, wvt, bq, bk, bv, qb, kb, vtb);

  k_attn<<<dim3(TT / 16, BB * HH), 32, 0, stream>>>(qb, kb, vtb, attn);

  k_oproj<<<dim3(BB * TT / 128, CC / 64), 128, 0, stream>>>(attn, wot, bo, out);
}
